// mlp_projection_filter_6433861009472
// MI455X (gfx1250) — matrix operand order repaired and checked by reading, not compiled, not run on hardware
//
#include <hip/hip_runtime.h>
#include <math.h>

// ---------------------------------------------------------------------------
// ADMM projection filter for MI455X (gfx1250, wave32, WMMA).
// kron(I6,.) structure: 414x414 KKT inverse -> one 69x69 inverse; all GEMMs
// are 64-wide per-DOF blocks. One fused kernel per ADMM iteration; each block
// owns a (32-row, 1-dof) subproblem in LDS. All WMMA operands are kept in
// fragment-native packed layout so every operand is one 32B LDS vector load.
// ---------------------------------------------------------------------------

typedef __attribute__((ext_vector_type(16))) _Float16 v16h;
typedef __attribute__((ext_vector_type(8)))  _Float16 v8h;
typedef __attribute__((ext_vector_type(8)))  float    v8f;

#define NUMR   128
#define NVS    64
#define NDOFS  6
#define BATCH  4096
#define MTILE  32
#define NWAVES 8
#define HDIM   96
#define NITER  15

// packed-constant pool layout (in halves)
#define PK_MB(cm)  ((cm) * 8192)            // Mc as B   (K=128,N=64):  16 tiles
#define PK_MBT(cm) (24576 + (cm) * 8192)    // Mc^T as B (K=64, N=128): 16 tiles
#define PK_HIB     49152                    // Hinv^T    (K=96, N=64):  12 tiles
#define PK_TOTAL   55296

__device__ __forceinline__ float red8(float v) {
  v += __shfl_xor(v, 1, 32);
  v += __shfl_xor(v, 2, 32);
  v += __shfl_xor(v, 4, 32);
  return v;
}

// A-fragment packed index: element (m, k) of an (MTILE x K) A matrix.
// Contiguous-k 16x32 f16 A layout (same k order as B pack): lane = h*16 + m%16, h=(kk>>4)&1,
// slot = kk&15. A contiguous group of 8 k's -> 8 contig halves.
__device__ __forceinline__ int apack_base(int KT, int m, int kg /* k/8 */) {
  int mt = m >> 4, q = m & 15;
  int kt = kg >> 2, g = kg & 3;
  return (((mt * KT + kt) * 32) + (g >> 1) * 16 + q) * 16 + (g & 1) * 8;
}

// Packed-operand cooperative GEMM: 8 waves split the 16x16 output tiles.
// Ap: A fragments [mt][kt][lane][16]; Bp: B fragments [kt][nt][lane][16].
template <int MT, int NT, int KT, typename FCL, typename FCS>
__device__ __forceinline__ void wmma_gemm_pk(const _Float16* __restrict__ Ap,
                                             const _Float16* __restrict__ Bp,
                                             FCL fcl, FCS fcs) {
  const int wave = threadIdx.x >> 5;
  const int lane = threadIdx.x & 31;
  const int h = lane >> 4, q = lane & 15;
  for (int t = wave; t < MT * NT; t += NWAVES) {
    const int mt = t / NT, nt = t % NT;
    v8f acc;
#pragma unroll
    for (int r = 0; r < 8; ++r) acc[r] = fcl((mt << 4) + r + 8 * h, (nt << 4) + q);
#pragma unroll
    for (int kt = 0; kt < KT; ++kt) {
      v16h a = *(const v16h*)(Ap + (((mt * KT + kt) * 32) + lane) * 16);
      v16h b = *(const v16h*)(Bp + (((kt * NT + nt) * 32) + lane) * 16);
      acc = __builtin_amdgcn_wmma_f32_16x16x32_f16(
          false, a, false, b, (short)0, acc, false, false);
    }
#pragma unroll
    for (int r = 0; r < 8; ++r) fcs((mt << 4) + r + 8 * h, (nt << 4) + q, acc[r]);
  }
}

// ---------------------------------------------------------------------------
// Prep: G = I + 2(P'P+Pd'Pd+Pdd'Pdd); H=[[G,E'],[E,0]] padded to 96 with I;
// Gauss-Jordan inverse in LDS; emit all B operands pre-swizzled into WMMA
// B-fragment order (f16).
// ---------------------------------------------------------------------------
__global__ __launch_bounds__(256) void pf_prep(
    const float* __restrict__ P, const float* __restrict__ Pdot,
    const float* __restrict__ Pddot, _Float16* __restrict__ Pk) {
  __shared__ float Hp[HDIM * HDIM];
  __shared__ float Iv[HDIM * HDIM];
  __shared__ float fcol[HDIM];
  __shared__ float piv;
  const int tid = threadIdx.x;
  for (int i = tid; i < HDIM * HDIM; i += 256) { Hp[i] = 0.f; Iv[i] = 0.f; }
  __syncthreads();
  for (int idx = tid; idx < 64 * 64; idx += 256) {
    int i = idx >> 6, j = idx & 63;
    float s = 0.f;
    for (int r = 0; r < NUMR; ++r)
      s += P[r * 64 + i] * P[r * 64 + j] + Pdot[r * 64 + i] * Pdot[r * 64 + j] +
           Pddot[r * 64 + i] * Pddot[r * 64 + j];
    Hp[i * HDIM + j] = ((i == j) ? 1.f : 0.f) + 2.f * s;  // RHO = 1
  }
  for (int idx = tid; idx < 5 * 64; idx += 256) {
    int k = idx / 64, j = idx % 64;
    float e = (k == 0) ? P[j] : (k == 1) ? Pdot[j] : (k == 2) ? Pddot[j]
            : (k == 3) ? Pdot[127 * 64 + j] : Pddot[127 * 64 + j];
    Hp[(64 + k) * HDIM + j] = e;
    Hp[j * HDIM + 64 + k] = e;
  }
  __syncthreads();
  for (int i = tid; i < HDIM; i += 256) {
    if (i >= 69) Hp[i * HDIM + i] = 1.f;
    Iv[i * HDIM + i] = 1.f;
  }
  __syncthreads();
  for (int k = 0; k < HDIM; ++k) {
    if (tid == 0) piv = 1.f / Hp[k * HDIM + k];
    __syncthreads();
    float pv = piv;
    for (int j = tid; j < HDIM; j += 256) {
      Hp[k * HDIM + j] *= pv; Iv[k * HDIM + j] *= pv;
    }
    __syncthreads();
    for (int i = tid; i < HDIM; i += 256) fcol[i] = (i == k) ? 0.f : Hp[i * HDIM + k];
    __syncthreads();
    for (int idx = tid; idx < HDIM * HDIM; idx += 256) {
      int i = idx / HDIM, j = idx % HDIM;
      if (i != k) {
        float f = fcol[i];
        Hp[idx] -= f * Hp[k * HDIM + j];
        Iv[idx] -= f * Iv[k * HDIM + j];
      }
    }
    __syncthreads();
  }
  // B-fragment swizzle: element (k,n) -> tile(kt,nt), lane=(kk>>4)*16+(n&15),
  // slot = kk&15.
  for (int cm = 0; cm < 3; ++cm) {
    const float* Msrc = (cm == 0) ? Pdot : (cm == 1) ? Pddot : P;
    for (int idx = tid; idx < NUMR * NVS; idx += 256) {   // Mc as B (128x64)
      int k = idx >> 6, n = idx & 63;
      int kt = k >> 5, kk = k & 31, nt = n >> 4;
      Pk[PK_MB(cm) + (((kt * 4 + nt) * 32) + (kk >> 4) * 16 + (n & 15)) * 16 +
         (kk & 15)] = (_Float16)Msrc[k * 64 + n];
    }
    for (int idx = tid; idx < 64 * 128; idx += 256) {     // Mc^T as B (64x128)
      int k = idx >> 7, n = idx & 127;
      int kt = k >> 5, kk = k & 31, nt = n >> 4;
      Pk[PK_MBT(cm) + (((kt * 8 + nt) * 32) + (kk >> 4) * 16 + (n & 15)) * 16 +
         (kk & 15)] = (_Float16)Msrc[n * 64 + k];
    }
  }
  for (int idx = tid; idx < HDIM * 64; idx += 256) {      // Hinv^T as B (96x64)
    int k = idx >> 6, n = idx & 63;
    int kt = k >> 5, kk = k & 31, nt = n >> 4;
    Pk[PK_HIB + (((kt * 4 + nt) * 32) + (kk >> 4) * 16 + (n & 15)) * 16 +
       (kk & 15)] = (_Float16)((k < 69) ? Iv[n * HDIM + k] : 0.f);
  }
}

// ---------------------------------------------------------------------------
// One fused ADMM iteration (iter>=0) or the s-initialization pass (iter=-1).
// grid = (BATCH/MTILE, NDOFS); block = 256 = 8 wave32.
// ---------------------------------------------------------------------------
__global__ __launch_bounds__(256) void pf_iter(
    const _Float16* __restrict__ Pk,    // packed B operands (PK_TOTAL halves)
    float* __restrict__ c_state,        // B x 384
    float* __restrict__ l_state,        // B x 1152 (v|a|p)
    _Float16* __restrict__ s_state,     // 3 x B x 6 x 256
    const float* __restrict__ b_eq,     // B x 30
    float* __restrict__ np,             // 15 x 10 x 6 x B partial sumsq
    int iter) {
  __shared__ __align__(32) _Float16 sPk[PK_TOTAL];        // all B fragments
  __shared__ __align__(32) _Float16 sDfA[8 * 512];        // A pack, K=128
  __shared__ __align__(32) _Float16 sLinA[6 * 512];       // A pack, K=96
  __shared__ __align__(32) _Float16 sPrA[4 * 512];        // A pack, K=64
  __shared__ __align__(32) float sC[MTILE * NVS];
  __shared__ __align__(32) float sLin[MTILE * NVS];
  __shared__ __align__(32) float sPr[MTILE * NVS];
  __shared__ __align__(32) float sY[MTILE * NUMR];
  __shared__ __align__(32) float sDl[MTILE * NVS];
  __shared__ float sBeq[MTILE * 5];
  __shared__ float sAcc[10 * MTILE];

  const int tid = threadIdx.x;
  const int dof = blockIdx.y;
  const int row0 = blockIdx.x * MTILE;

  if (tid < MTILE) {  // warm L2/L0 for the per-row state we touch soon
    size_t r = (size_t)(row0 + tid);
    __builtin_prefetch(&l_state[r * 1152 + dof * 64], 0, 3);
    __builtin_prefetch(&s_state[(r * NDOFS + dof) * 256], 0, 3);
    __builtin_prefetch(&s_state[(size_t)BATCH * NDOFS * 256 + (r * NDOFS + dof) * 256], 0, 3);
    __builtin_prefetch(&s_state[(size_t)2 * BATCH * NDOFS * 256 + (r * NDOFS + dof) * 256], 0, 3);
  }
  for (int i = tid; i < PK_TOTAL / 8; i += 256)
    ((v8h*)sPk)[i] = ((const v8h*)Pk)[i];
  for (int i = tid; i < MTILE * NVS; i += 256) {
    int m = i >> 6, j = i & 63;
    sC[i] = c_state[(size_t)(row0 + m) * 384 + dof * 64 + j];
  }
  for (int i = tid; i < MTILE * 5; i += 256)
    sBeq[i] = b_eq[(row0 + i / 5) * 30 + dof * 5 + (i % 5)];
  for (int i = tid; i < 10 * MTILE; i += 256) sAcc[i] = 0.f;
  __syncthreads();

  if (iter >= 0) {
    // rhs = -lincost = l_v + l_a + l_p + c + sum_c (s2 - s1) @ Mc    (RHO=1)
    for (int i = tid; i < MTILE * NVS; i += 256) {
      int m = i >> 6, j = i & 63;
      size_t lb = (size_t)(row0 + m) * 1152 + dof * 64 + j;
      sLin[i] = l_state[lb] + l_state[lb + 384] + l_state[lb + 768] + sC[i];
    }
    __syncthreads();
    for (int cm = 0; cm < 3; ++cm) {
      const _Float16* sp = s_state + (size_t)cm * BATCH * NDOFS * 256;
      // pack A = (s2 - s1) into fragment order: 32 rows x 16 chunks of 8
#pragma unroll
      for (int ii = 0; ii < 2; ++ii) {
        int cc = tid + ii * 256;
        int m = cc >> 4, kg = cc & 15;
        size_t sb = ((size_t)(row0 + m) * NDOFS + dof) * 256 + kg * 8;
        v8h s1 = *(const v8h*)(sp + sb);
        v8h s2 = *(const v8h*)(sp + sb + 128);
        v8h d = s2 - s1;
        *(v8h*)(sDfA + apack_base(4, m, kg)) = d;
      }
      __syncthreads();
      wmma_gemm_pk<2, 4, 4>(sDfA, sPk + PK_MB(cm),
        [&](int m, int n) -> float { return sLin[m * NVS + n]; },
        [&](int m, int n, float v) { sLin[m * NVS + n] = v; });
      __syncthreads();
    }
    // pack A = [rhs | b_eq | 0] (K = 96) and solve: primal = A @ Hinv_p^T
    for (int i = tid; i < MTILE * 12; i += 256) {
      int m = i / 12, kg = i % 12;
      v8h v;
#pragma unroll
      for (int e = 0; e < 8; ++e) {
        int k = kg * 8 + e;
        float x = (k < 64) ? sLin[m * NVS + k]
                           : ((k < 69) ? sBeq[m * 5 + (k - 64)] : 0.f);
        v[e] = (_Float16)x;
      }
      *(v8h*)(sLinA + apack_base(3, m, kg)) = v;
    }
    __syncthreads();
    wmma_gemm_pk<2, 4, 3>(sLinA, sPk + PK_HIB,
      [&](int, int) -> float { return 0.f; },
      [&](int m, int n, float v) { sPr[m * NVS + n] = v; });
    __syncthreads();
    // ||primal - c||^2 (term 9) + pack primal as A (K = 64)
    {
      int m = tid >> 3, kg = tid & 7;
      float a = 0.f;
      v8h ph;
#pragma unroll
      for (int e = 0; e < 8; ++e) {
        float p = sPr[m * NVS + kg * 8 + e];
        float d = p - sC[m * NVS + kg * 8 + e];
        a += d * d;
        ph[e] = (_Float16)p;
      }
      *(v8h*)(sPrA + apack_base(2, m, kg)) = ph;
      a = red8(a);
      if ((tid & 7) == 0) sAcc[9 * MTILE + m] += a;
    }
    __syncthreads();
  } else {
    int m = tid >> 3, kg = tid & 7;   // init: A pack of c_samples
    v8h ph;
#pragma unroll
    for (int e = 0; e < 8; ++e) ph[e] = (_Float16)sC[m * NVS + kg * 8 + e];
    *(v8h*)(sPrA + apack_base(2, m, kg)) = ph;
    __syncthreads();
  }

  for (int cm = 0; cm < 3; ++cm) {
    const float bc = (cm == 0) ? 0.8f : (cm == 1) ? 1.8f : 3.14159265358979f;
    _Float16* sp = s_state + (size_t)cm * BATCH * NDOFS * 256;
    // y = primal @ Mc^T   (Ax = [y, -y])
    wmma_gemm_pk<2, 8, 2>(sPrA, sPk + PK_MBT(cm),
      [&](int, int) -> float { return 0.f; },
      [&](int m, int n, float v) { sY[m * NUMR + n] = v; });
    __syncthreads();
    int m = tid >> 3;
    if (iter >= 0) {
      float aS = 0.f, aR = 0.f;
#pragma unroll
      for (int ii = 0; ii < 2; ++ii) {
        int kg = (2 * tid + ii) & 15;         // threads 8r..8r+7 own row r
        int j0 = kg * 8;
        size_t sb = ((size_t)(row0 + m) * NDOFS + dof) * 256 + j0;
        v8h o1 = *(const v8h*)(sp + sb);
        v8h o2 = *(const v8h*)(sp + sb + 128);
        v8h n1, n2, rd;
#pragma unroll
        for (int e = 0; e < 8; ++e) {
          float y = sY[m * NUMR + j0 + e];
          float s1 = fmaxf(0.f, bc - y), s2 = fmaxf(0.f, bc + y);
          float r1 = fmaxf(y - bc, 0.f), r2 = fmaxf(-y - bc, 0.f);
          float d1 = s1 - (float)o1[e], d2 = s2 - (float)o2[e];
          aS += d1 * d1 + d2 * d2;
          aR += r1 * r1 + r2 * r2;
          n1[e] = (_Float16)s1; n2[e] = (_Float16)s2;
          rd[e] = (_Float16)(r1 - r2);
        }
        *(v8h*)(sp + sb) = n1;
        *(v8h*)(sp + sb + 128) = n2;
        *(v8h*)(sDfA + apack_base(4, m, kg)) = rd;   // A for dl GEMM
      }
      aS = red8(aS); aR = red8(aR);
      if ((tid & 7) == 0) { sAcc[(6 + cm) * MTILE + m] += aS; sAcc[cm * MTILE + m] += aR; }
      __syncthreads();
      // dl = (r1 - r2) @ Mc ; l -= dl
      wmma_gemm_pk<2, 4, 4>(sDfA, sPk + PK_MB(cm),
        [&](int, int) -> float { return 0.f; },
        [&](int mm, int n, float v) { sDl[mm * NVS + n] = v; });
      __syncthreads();
      {
        int kg = tid & 7, j0 = kg * 8;
        float aL = 0.f;
        size_t lb = (size_t)(row0 + m) * 1152 + cm * 384 + dof * 64 + j0;
#pragma unroll
        for (int e = 0; e < 8; ++e) {
          float d = sDl[m * NVS + j0 + e];
          l_state[lb + e] = l_state[lb + e] - d;
          aL += d * d;
        }
        aL = red8(aL);
        if ((tid & 7) == 0) sAcc[(3 + cm) * MTILE + m] += aL;
      }
      __syncthreads();
    } else {
#pragma unroll
      for (int ii = 0; ii < 2; ++ii) {
        int kg = (2 * tid + ii) & 15;
        int j0 = kg * 8;
        size_t sb = ((size_t)(row0 + m) * NDOFS + dof) * 256 + j0;
        v8h n1, n2;
#pragma unroll
        for (int e = 0; e < 8; ++e) {
          float y = sY[m * NUMR + j0 + e];
          n1[e] = (_Float16)fmaxf(0.f, bc - y);
          n2[e] = (_Float16)fmaxf(0.f, bc + y);
        }
        *(v8h*)(sp + sb) = n1;
        *(v8h*)(sp + sb + 128) = n2;
      }
      __syncthreads();
    }
  }

  if (iter >= 0) {
    for (int i = tid; i < MTILE * NVS; i += 256) {
      int m = i >> 6, j = i & 63;
      c_state[(size_t)(row0 + m) * 384 + dof * 64 + j] = sPr[i];
    }
    for (int i = tid; i < 10 * MTILE; i += 256) {
      int term = i >> 5, mm = i & 31;
      np[(((size_t)iter * 10 + term) * NDOFS + dof) * BATCH + (row0 + mm)] =
          sAcc[term * MTILE + mm];
    }
  }
}

__global__ void pf_zero(float* __restrict__ p, int n) {
  int i = blockIdx.x * 256 + threadIdx.x;
  if (i < n) p[i] = 0.f;
}

__global__ void pf_final(const float* __restrict__ np, float* __restrict__ out) {
  int row = blockIdx.x * 256 + threadIdx.x;
  if (row >= BATCH) return;
  float af = 0.f, ap = 0.f;
  for (int it = 0; it < NITER; ++it) {
    for (int term = 0; term < 10; ++term) {
      float s = 0.f;
      for (int d = 0; d < NDOFS; ++d)
        s += np[(((size_t)it * 10 + term) * NDOFS + d) * BATCH + row];
      float r = sqrtf(s);
      if (term < 3) ap += r; else af += r;
    }
  }
  out[row] = af / (float)NITER;          // acc_res_fixed
  out[BATCH + row] = ap / (float)NITER;  // acc_res_primal
}

extern "C" void kernel_launch(void* const* d_in, const int* in_sizes, int n_in,
                              void* d_out, int out_size, void* d_ws, size_t ws_size,
                              hipStream_t stream) {
  (void)in_sizes; (void)n_in; (void)out_size; (void)ws_size;
  const float* P     = (const float*)d_in[0];
  const float* Pdot  = (const float*)d_in[1];
  const float* Pddot = (const float*)d_in[2];
  const float* lam   = (const float*)d_in[3];
  const float* csamp = (const float*)d_in[4];
  const float* beq   = (const float*)d_in[5];

  char* w = (char*)d_ws;
  _Float16* Pk = (_Float16*)w;      w += (size_t)PK_TOTAL * sizeof(_Float16);
  float* c_state = (float*)w;       w += (size_t)BATCH * 384 * sizeof(float);
  float* l_state = (float*)w;       w += (size_t)BATCH * 1152 * sizeof(float);
  _Float16* s_state = (_Float16*)w; w += (size_t)3 * BATCH * NDOFS * 256 * sizeof(_Float16);
  float* np = (float*)w;            w += (size_t)NITER * 10 * NDOFS * BATCH * sizeof(float);

  pf_prep<<<1, 256, 0, stream>>>(P, Pdot, Pddot, Pk);
  int nz = NITER * 10 * NDOFS * BATCH;
  pf_zero<<<(nz + 255) / 256, 256, 0, stream>>>(np, nz);
  hipMemcpyAsync(c_state, csamp, (size_t)BATCH * 384 * sizeof(float),
                 hipMemcpyDeviceToDevice, stream);
  hipMemcpyAsync(l_state, lam, (size_t)BATCH * 1152 * sizeof(float),
                 hipMemcpyDeviceToDevice, stream);
  dim3 grid(BATCH / MTILE, NDOFS);
  pf_iter<<<grid, 256, 0, stream>>>(Pk, c_state, l_state, s_state, beq, np, -1);
  for (int it = 0; it < NITER; ++it)
    pf_iter<<<grid, 256, 0, stream>>>(Pk, c_state, l_state, s_state, beq, np, it);
  hipMemcpyAsync(d_out, c_state, (size_t)BATCH * 384 * sizeof(float),
                 hipMemcpyDeviceToDevice, stream);
  pf_final<<<(BATCH + 255) / 256, 256, 0, stream>>>(np, (float*)d_out + (size_t)BATCH * 384);
}
